// Attention_49254684950736
// MI455X (gfx1250) — matrix operand order repaired and checked by reading, not compiled, not run on hardware
//
#include <hip/hip_runtime.h>
#include <hip/hip_bf16.h>
#include <math.h>

// ---------------------------------------------------------------------------
// FAVOR+ (Performer) causal chunked attention for MI455X (gfx1250, wave32).
// All GEMMs run on v_wmma_f32_16x16x32_f16. Shapes fixed by the reference:
//   B=2, H=16, L=4096, D=M=64, chunk=128, NC=32.
// Fragment operands are fed by 128-bit loads only:
//   - A / B-transposed fragments: row-contiguous -> 2x ds_load_b128 per lane
//   - B fragments: operands staged in "fragment order" -> 2x b128 per lane
// DATA_NORM is folded into the proj staging so WMMA accumulators are stored
// without a post-scale (no v_pk_mul chains / hazard NOPs after the WMMAs).
// ---------------------------------------------------------------------------

typedef __attribute__((ext_vector_type(16))) _Float16 v16h;
typedef __attribute__((ext_vector_type(8)))  _Float16 v8h;
typedef __attribute__((ext_vector_type(4)))  _Float16 v4h;
typedef __attribute__((ext_vector_type(8)))  float    v8f;
typedef __attribute__((ext_vector_type(4)))  float    v4f;

#define NB  2
#define NH  16
#define NL  4096
#define ND  64          // head dim == feature dim M
#define NCH 128         // chunk size
#define NCC 32          // number of chunks (= L / chunk) == token blocks of 128
#define DATA_NORM 0.3535533905932738f   // 64^-0.25
#define DN2       0.125f                // 64^-0.5
#define RATIO     0.125f                // 64^-0.5
#define EPSV      1e-4f

static __device__ __forceinline__ v8f wmma16(v16h a, v16h b, v8f c) {
  // D = A(16x32 f16) * B(32x16 f16) + C(16x16 f32)
  return __builtin_amdgcn_wmma_f32_16x16x32_f16(false, a, false, b,
                                                (short)0, c, false, false);
}

static __device__ __forceinline__ v16h cat8(v8h lo, v8h hi) {
  return __builtin_shufflevector(lo, hi, 0, 1, 2, 3, 4, 5, 6, 7,
                                 8, 9, 10, 11, 12, 13, 14, 15);
}

// A fragment, logical A[16][32] row-major at `base` with row stride ld (halves).
// Lane l<16: row=l, K 0..15; lane l>=16: row=l-16, K 16..31.
static __device__ __forceinline__ v16h load_a_frag(const _Float16* base, int ld) {
  const int lane = threadIdx.x & 31;
  const _Float16* p = base + (lane & 15) * ld + ((lane >> 4) << 4);
  return cat8(*(const v8h*)(p), *(const v8h*)(p + 8));
}

// B fragment from a transposed view: logical B[k][n] = base[n*ld + k].
static __device__ __forceinline__ v16h load_bt_frag(const _Float16* base, int ld) {
  const int lane = threadIdx.x & 31;
  const _Float16* p = base + (lane & 15) * ld + ((lane >> 4) << 4);
  return cat8(*(const v8h*)(p), *(const v8h*)(p + 8));
}

// B fragment from fragment-ordered storage: tile base -> lane*16 halves.
static __device__ __forceinline__ v16h load_b_frag_ord(const _Float16* tile_base) {
  const _Float16* p = tile_base + (threadIdx.x & 31) * 16;
  return cat8(*(const v8h*)(p), *(const v8h*)(p + 8));
}

// Fragment-order scatter offset for a B operand element (kk, n):
// tile = (kk/32)*4 + n/16 ; lane = ((kk>>4)&1)*16 + (n&15) ; j = kk&15
static __device__ __forceinline__ int bfrag_off(int kk, int n) {
  return (((kk >> 5) * 4 + (n >> 4)) * 512) +
         (((((kk >> 4) & 1) << 4) + (n & 15)) << 4) + (kk & 15);
}

// C/D layout: lane l<16 -> col=l, VGPR r -> row r; lane l>=16 -> col=l-16, row 8+r.
static __device__ __forceinline__ void store_c_f32(float* dst, int ld, v8f acc) {
  const int lane  = threadIdx.x & 31;
  const int col   = lane & 15;
  const int rbase = (lane >> 4) * 8;
#pragma unroll
  for (int r = 0; r < 8; ++r) dst[(rbase + r) * ld + col] = acc[r];
}

// ---------------------------------------------------------------------------
// Shared device routine: stage a 128x64 f32 tile (row-major f16) + 64x64 proj
// (fragment order, pre-scaled by DATA_NORM) into LDS;
// dd = x @ (proj*DATA_NORM) into f32 LDS.
// ---------------------------------------------------------------------------
static __device__ __forceinline__ void feat_gemm(const float* __restrict__ xg,
                                                 const float* __restrict__ projg,
                                                 _Float16* xLDS, _Float16* pLDS,
                                                 float* ddLDS) {
  const int tid = threadIdx.x;
  // x: vector f32->f16 convert, row-major (A operand)
  for (int i = tid * 4; i < 128 * 64; i += 256 * 4) {
    v4f w = *(const v4f*)(xg + i);
    v4h h; h[0] = (_Float16)w[0]; h[1] = (_Float16)w[1];
    h[2] = (_Float16)w[2]; h[3] = (_Float16)w[3];
    *(v4h*)(xLDS + i) = h;
  }
  // proj: coalesced read, DATA_NORM folded in, fragment-order scatter (K=64)
  for (int i = tid; i < 64 * 64; i += 256) {
    const int r = i >> 6, d = i & 63;
    pLDS[bfrag_off(r, d)] = (_Float16)(projg[i] * DATA_NORM);
  }
  __syncthreads();
  const int ti = tid >> 5;                       // wave id == 16-row strip
#pragma unroll
  for (int tc = 0; tc < 4; ++tc) {
    v8f acc = {};
#pragma unroll
    for (int ko = 0; ko < 64; ko += 32) {
      v16h a = load_a_frag(xLDS + (ti * 16) * 64 + ko, 64);
      v16h b = load_b_frag_ord(pLDS + ((ko >> 5) * 4 + tc) * 512);
      acc = wmma16(a, b, acc);
    }
    store_c_f32(ddLDS + (ti * 16) * 64 + tc * 16, 64, acc);
  }
  __syncthreads();
}

// Finalize helper: thread pair (t, t^1) owns one row; each thread handles 32
// columns; ss / row-max combined with one __shfl_xor.
static __device__ __forceinline__ void feat_finalize(const float* __restrict__ xrow,
                                                     const float* __restrict__ dd,
                                                     _Float16* __restrict__ orow,
                                                     float stab_in, bool use_row_max) {
  float ss = 0.f, mx = -1e30f;
#pragma unroll 8
  for (int j = 0; j < 32; ++j) {
    const float w = xrow[j];
    ss += w * w;
    mx = fmaxf(mx, dd[j]);
  }
  ss += __shfl_xor(ss, 1, 32);
  mx = fmaxf(mx, __shfl_xor(mx, 1, 32));
  const float st   = use_row_max ? mx : stab_in;
  const float diag = 0.5f * ss * DN2;
#pragma unroll
  for (int j = 0; j < 32; j += 4) {
    v4h o;
#pragma unroll
    for (int e = 0; e < 4; ++e)
      o[e] = (_Float16)(RATIO * (__expf(dd[j + e] - diag - st) + EPSV));
    *(v4h*)(orow + j) = o;
  }
}

// ---------------------------------------------------------------------------
// Kernel 1: q features (per-token stab), qp (f16) out. grid (NCC,NH,NB) x 256.
// ---------------------------------------------------------------------------
__global__ void __launch_bounds__(256)
feat_q_kernel(const float* __restrict__ q, const float* __restrict__ proj,
              _Float16* __restrict__ qp) {
  __shared__ _Float16 xLDS[128 * 64];
  __shared__ _Float16 pLDS[64 * 64];
  __shared__ float    ddLDS[128 * 64];
  const int    bh   = blockIdx.z * NH + blockIdx.y;
  const size_t tok0 = (size_t)bh * NL + (size_t)blockIdx.x * 128;
  feat_gemm(q + tok0 * ND, proj, xLDS, pLDS, ddLDS);
  const int t    = threadIdx.x;
  const int row  = t >> 1;
  const int half = (t & 1) * 32;
  feat_finalize(q + (tok0 + row) * ND + half,
                ddLDS + row * ND + half,
                qp + (tok0 + row) * ND + half,
                0.f, /*use_row_max=*/true);
}

// ---------------------------------------------------------------------------
// Kernel 2: per-block max of k's data_dash (toward global per-(b,h) stab).
// ---------------------------------------------------------------------------
__global__ void __launch_bounds__(256)
kmax_kernel(const float* __restrict__ k, const float* __restrict__ proj,
            float* __restrict__ blockmax) {
  __shared__ _Float16 xLDS[128 * 64];
  __shared__ _Float16 pLDS[64 * 64];
  __shared__ float    ddLDS[128 * 64];
  __shared__ float    red[256];
  const int    bh   = blockIdx.z * NH + blockIdx.y;
  const size_t tok0 = (size_t)bh * NL + (size_t)blockIdx.x * 128;
  feat_gemm(k + tok0 * ND, proj, xLDS, pLDS, ddLDS);
  const int t  = threadIdx.x;
  float     mx = -1e30f;
  for (int i = t; i < 128 * 64; i += 256) mx = fmaxf(mx, ddLDS[i]);
  red[t] = mx;
  __syncthreads();
  for (int s = 128; s > 0; s >>= 1) {
    if (t < s) red[t] = fmaxf(red[t], red[t + s]);
    __syncthreads();
  }
  if (t == 0) blockmax[bh * NCC + blockIdx.x] = red[0];
}

__global__ void __launch_bounds__(32)
kmax_reduce_kernel(const float* __restrict__ blockmax, float* __restrict__ stab) {
  const int t = threadIdx.x;       // one thread per (b,h), 32 total
  if (t < NB * NH) {
    float mx = -1e30f;
    for (int i = 0; i < NCC; ++i) mx = fmaxf(mx, blockmax[t * NCC + i]);
    stab[t] = mx;
  }
}

// ---------------------------------------------------------------------------
// Kernel 3: k features with global stab, kp (f16) out.
// ---------------------------------------------------------------------------
__global__ void __launch_bounds__(256)
feat_k_kernel(const float* __restrict__ k, const float* __restrict__ proj,
              const float* __restrict__ stab, _Float16* __restrict__ kp) {
  __shared__ _Float16 xLDS[128 * 64];
  __shared__ _Float16 pLDS[64 * 64];
  __shared__ float    ddLDS[128 * 64];
  const int    bh   = blockIdx.z * NH + blockIdx.y;
  const size_t tok0 = (size_t)bh * NL + (size_t)blockIdx.x * 128;
  feat_gemm(k + tok0 * ND, proj, xLDS, pLDS, ddLDS);
  const int t    = threadIdx.x;
  const int row  = t >> 1;
  const int half = (t & 1) * 32;
  feat_finalize(k + (tok0 + row) * ND + half,
                ddLDS + row * ND + half,
                kp + (tok0 + row) * ND + half,
                stab[bh], /*use_row_max=*/false);
}

// ---------------------------------------------------------------------------
// Kernel 4: per-chunk kv = kc^T @ vc (64x64, K=128) and ks = colsum(kc).
// kc staged transposed (A contiguous), vc staged fragment-order (B).
// ---------------------------------------------------------------------------
__global__ void __launch_bounds__(256)
chunk_kv_kernel(const _Float16* __restrict__ kp, const float* __restrict__ v,
                float* __restrict__ kvg, float* __restrict__ ksg) {
  __shared__ _Float16 kT[64 * 128];      // kT[m][j] = kp[j][m]
  __shared__ _Float16 vLDS[128 * 64];    // fragment order, K=tokens
  const int    bh   = blockIdx.z * NH + blockIdx.y;
  const int    c    = blockIdx.x;
  const size_t tok0 = (size_t)bh * NL + (size_t)c * NCH;
  const int t = threadIdx.x;
  {
    const _Float16* ks = kp + tok0 * ND;
    const float*    vs = v + tok0 * ND;
    for (int i = t; i < 128 * 64; i += 256) {
      const int r = i >> 6, d = i & 63;            // token, feature/dim
      kT[d * 128 + r]          = ks[i];            // transpose scatter
      vLDS[bfrag_off(r, d)]    = (_Float16)vs[i];  // frag-order scatter
    }
  }
  __syncthreads();
  float* kvc = kvg + ((size_t)bh * NCC + c) * (ND * ND);
  const int wid = t >> 5;
#pragma unroll
  for (int i = 0; i < 2; ++i) {                  // 16 tiles / 8 waves
    const int tt  = wid * 2 + i;
    const int tm  = tt >> 2;                     // m strip
    const int tdc = tt & 3;                      // d column tile
    v8f acc = {};
#pragma unroll
    for (int jo = 0; jo < 128; jo += 32) {       // K over tokens in chunk
      v16h a = load_a_frag(kT + (tm * 16) * 128 + jo, 128);
      v16h b = load_b_frag_ord(vLDS + ((jo >> 5) * 4 + tdc) * 512);
      acc = wmma16(a, b, acc);
    }
    store_c_f32(kvc + (tm * 16) * ND + tdc * 16, ND, acc);
  }
  if (t < ND) {
    float s = 0.f;
    for (int j = 0; j < NCH; ++j) s += (float)kT[t * 128 + j];
    ksg[((size_t)bh * NCC + c) * ND + t] = s;
  }
}

// ---------------------------------------------------------------------------
// Kernel 5: exclusive prefix over chunks. ks in-place (f32); kv prefix is
// emitted as f16 *fragment-ordered* kvh so out_kernel B-frags are two
// coalesced 16B global loads per lane.
// ---------------------------------------------------------------------------
__global__ void __launch_bounds__(256)
prefix_kernel(const float* __restrict__ kvg, float* __restrict__ ksg,
              _Float16* __restrict__ kvh) {
  const int bh = blockIdx.x;
  const int t  = threadIdx.x;
  const int m  = t >> 2;            // kv row (feature index)
  const int tc = t & 3;             // d block
  float run[16];
#pragma unroll
  for (int e = 0; e < 16; ++e) run[e] = 0.f;
  for (int c = 0; c < NCC; ++c) {
    const float* p = kvg + ((size_t)bh * NCC + c) * (ND * ND) + m * ND + tc * 16;
    _Float16*    o = kvh + ((size_t)bh * NCC + c) * (8 * 512);
#pragma unroll
    for (int e = 0; e < 16; ++e) {
      o[bfrag_off(m, tc * 16 + e)] = (_Float16)run[e];   // exclusive prefix
      run[e] += p[e];
    }
  }
  if (t < ND) {
    float r = 0.f;
    for (int c = 0; c < NCC; ++c) {
      float* p = ksg + ((size_t)bh * NCC + c) * ND + t;
      float tmp = *p;
      *p = r;
      r += tmp;
    }
  }
}

// ---------------------------------------------------------------------------
// Kernel 6: masked scores S = tril(qc @ kc^T) (f16 out) + full denominator
//           den = rowsum(S) + qc . ks_prev.
// ---------------------------------------------------------------------------
__global__ void __launch_bounds__(256)
scores_kernel(const _Float16* __restrict__ qp, const _Float16* __restrict__ kp,
              const float* __restrict__ ksg, _Float16* __restrict__ Sg,
              float* __restrict__ deng) {
  __shared__ _Float16 qLDS[128 * 64];
  __shared__ _Float16 kLDS[128 * 64];
  __shared__ float    denLDS[128];
  __shared__ float    kspLDS[64];
  const int    bh   = blockIdx.z * NH + blockIdx.y;
  const int    c    = blockIdx.x;
  const size_t tok0 = (size_t)bh * NL + (size_t)c * NCH;
  const int t = threadIdx.x;
  {
    const v8h* qs = (const v8h*)(qp + tok0 * ND);
    const v8h* ks = (const v8h*)(kp + tok0 * ND);
    v8h* qd = (v8h*)qLDS;
    v8h* kd = (v8h*)kLDS;
    for (int i = t; i < (128 * 64) / 8; i += 256) { qd[i] = qs[i]; kd[i] = ks[i]; }
    if (t < ND) kspLDS[t] = ksg[((size_t)bh * NCC + c) * ND + t];
  }
  __syncthreads();

  const int ti    = t >> 5;                      // wave id == row strip
  const int lane  = t & 31;
  const int col   = lane & 15;
  const int rbase = (lane >> 4) * 8;
  _Float16* Srow = Sg + tok0 * NCH;              // chunk-local 128x128
  float dpart[8];
#pragma unroll
  for (int r = 0; r < 8; ++r) dpart[r] = 0.f;

  for (int tj = 0; tj < 8; ++tj) {
    if (tj > ti) {                               // strictly-future tiles: zeros
#pragma unroll
      for (int r = 0; r < 8; ++r)
        Srow[(ti * 16 + rbase + r) * NCH + tj * 16 + col] = (_Float16)0.f;
    } else {
      v8f acc = {};
#pragma unroll
      for (int ko = 0; ko < 64; ko += 32) {      // K over feature dim m
        v16h a = load_a_frag(qLDS + (ti * 16) * 64 + ko, 64);
        v16h b = load_bt_frag(kLDS + (tj * 16) * 64 + ko, 64); // B[m][j]=k[j][m]
        acc = wmma16(a, b, acc);
      }
      const bool diag = (tj == ti);
#pragma unroll
      for (int r = 0; r < 8; ++r) {
        const int row = rbase + r;
        float w = acc[r];
        if (diag && col > row) w = 0.f;          // causal mask within chunk
        dpart[r] += w;
        Srow[(ti * 16 + row) * NCH + tj * 16 + col] = (_Float16)w;
      }
    }
  }
  // reduce rowsums across the 16 column-lanes of each half-wave
#pragma unroll
  for (int m = 1; m < 16; m <<= 1) {
#pragma unroll
    for (int r = 0; r < 8; ++r) dpart[r] += __shfl_xor(dpart[r], m, 32);
  }
  if (lane == 0) {
#pragma unroll
    for (int r = 0; r < 8; ++r) denLDS[ti * 16 + r] = dpart[r];
  }
  if (lane == 16) {
#pragma unroll
    for (int r = 0; r < 8; ++r) denLDS[ti * 16 + 8 + r] = dpart[r];
  }
  __syncthreads();
  if (t < 128) {
    const v8h* qr = (const v8h*)(qLDS + t * 64);
    float id = 0.f;
#pragma unroll
    for (int g = 0; g < 8; ++g) {
      v8h x = qr[g];
#pragma unroll
      for (int e = 0; e < 8; ++e) id += (float)x[e] * kspLDS[g * 8 + e];
    }
    deng[tok0 + t] = denLDS[t] + id;
  }
}

// ---------------------------------------------------------------------------
// Kernel 7: out = (S @ vc + qc @ kv_prev) / den.
// 6 chained WMMAs per 16x16 output tile; S read as A-frags straight from
// global (contiguous 16B runs); kv_prev read as fragment-ordered B from ws.
// ---------------------------------------------------------------------------
__global__ void __launch_bounds__(256)
out_kernel(const _Float16* __restrict__ qp, const float* __restrict__ v,
           const _Float16* __restrict__ Sg, const _Float16* __restrict__ kvh,
           const float* __restrict__ deng, float* __restrict__ out) {
  __shared__ _Float16 qLDS[128 * 64];
  __shared__ _Float16 vLDS[128 * 64];            // fragment order
  const int    bh   = blockIdx.z * NH + blockIdx.y;
  const int    c    = blockIdx.x;
  const size_t tok0 = (size_t)bh * NL + (size_t)c * NCH;
  const int t = threadIdx.x;
  {
    const v8h* qs = (const v8h*)(qp + tok0 * ND);
    v8h*       qd = (v8h*)qLDS;
    for (int i = t; i < (128 * 64) / 8; i += 256) qd[i] = qs[i];
    const float* vs = v + tok0 * ND;
    for (int i = t; i < 128 * 64; i += 256) {
      const int r = i >> 6, d = i & 63;
      vLDS[bfrag_off(r, d)] = (_Float16)vs[i];
    }
    __builtin_prefetch(Sg + tok0 * NCH, 0, 1);   // warm S before the GEMM phase
  }
  __syncthreads();

  const int       ti   = t >> 5;
  const int       lane = t & 31;
  const int       col  = lane & 15;
  const int       rb   = (lane >> 4) * 8;
  const _Float16* kvc  = kvh + ((size_t)bh * NCC + c) * (8 * 512);
  const _Float16* Sst  = Sg + (tok0 + ti * 16) * NCH;   // this wave's S strip
  const float*    dd   = deng + tok0 + ti * 16;
  float*          orow = out + (tok0 + ti * 16) * ND;

#pragma unroll
  for (int tc = 0; tc < 4; ++tc) {
    v8f acc = {};
#pragma unroll
    for (int ko = 0; ko < 64; ko += 32) {        // inter: qc @ kv_prev
      v16h a = load_a_frag(qLDS + (ti * 16) * 64 + ko, 64);
      v16h b = load_b_frag_ord(kvc + ((ko >> 5) * 4 + tc) * 512);
      acc = wmma16(a, b, acc);
    }
#pragma unroll
    for (int jo = 0; jo < 128; jo += 32) {       // intra: S @ vc
      v16h a = load_a_frag(Sst + jo, NCH);
      v16h b = load_b_frag_ord(vLDS + ((jo >> 5) * 4 + tc) * 512);
      acc = wmma16(a, b, acc);
    }
#pragma unroll
    for (int r = 0; r < 8; ++r) {
      const int row = rb + r;
      orow[row * ND + tc * 16 + col] = acc[r] / dd[row];
    }
  }
}

// ---------------------------------------------------------------------------
// Host-side launcher. Inputs: q, k, v, projection_matrix, chunk_size.
// Workspace carved from d_ws (~93 MB total).
// ---------------------------------------------------------------------------
extern "C" void kernel_launch(void* const* d_in, const int* in_sizes, int n_in,
                              void* d_out, int out_size, void* d_ws, size_t ws_size,
                              hipStream_t stream) {
  (void)in_sizes; (void)n_in; (void)out_size; (void)ws_size;
  const float* q    = (const float*)d_in[0];
  const float* k    = (const float*)d_in[1];
  const float* v    = (const float*)d_in[2];
  const float* proj = (const float*)d_in[3];
  float*       o    = (float*)d_out;

  char*  ws  = (char*)d_ws;
  size_t off = 0;
  auto carve = [&](size_t bytes) -> char* {
    char* p = ws + off;
    off = (off + bytes + 255) & ~(size_t)255;
    return p;
  };
  const size_t BHL = (size_t)NB * NH * NL;
  _Float16* qp       = (_Float16*)carve(BHL * ND * sizeof(_Float16));        // 16.8 MB
  _Float16* kp       = (_Float16*)carve(BHL * ND * sizeof(_Float16));        // 16.8 MB
  _Float16* Sws      = (_Float16*)carve(BHL * NCH * sizeof(_Float16));       // 33.6 MB
  float*    kvg      = (float*)carve((size_t)NB * NH * NCC * ND * ND * 4);   // 16.8 MB
  _Float16* kvh      = (_Float16*)carve((size_t)NB * NH * NCC * ND * ND * 2);// 8.4 MB
  float*    ksg      = (float*)carve((size_t)NB * NH * NCC * ND * 4);        // 0.26 MB
  float*    deng     = (float*)carve(BHL * 4);                               // 0.5 MB
  float*    blockmax = (float*)carve((size_t)NB * NH * NCC * 4);
  float*    stab     = (float*)carve((size_t)NB * NH * 4);

  const dim3 blk(256);
  const dim3 grd(NCC, NH, NB);   // 32 token-blocks/chunks x 16 heads x 2 batch

  feat_q_kernel      <<<grd, blk, 0, stream>>>(q, proj, qp);
  kmax_kernel        <<<grd, blk, 0, stream>>>(k, proj, blockmax);
  kmax_reduce_kernel <<<dim3(1), dim3(32), 0, stream>>>(blockmax, stab);
  feat_k_kernel      <<<grd, blk, 0, stream>>>(k, proj, stab, kp);
  chunk_kv_kernel    <<<grd, blk, 0, stream>>>(kp, v, kvg, ksg);
  prefix_kernel      <<<dim3(NB * NH), blk, 0, stream>>>(kvg, ksg, kvh);
  scores_kernel      <<<grd, blk, 0, stream>>>(qp, kp, ksg, Sws, deng);
  out_kernel         <<<grd, blk, 0, stream>>>(qp, v, Sws, kvh, deng, o);
}
